// OriginRTDETRDecoder_84782654423148
// MI455X (gfx1250) — matrix operand order repaired and checked by reading, not compiled, not run on hardware
//
#include <hip/hip_runtime.h>
#include <cstddef>
#include <cstdint>

// ---------------- model constants ----------------
#define D_      256
#define NHEAD_  8
#define DH_     32
#define NQ_     300
#define BS_     16
#define LM_     8400
#define NLEV_   3
#define NPTS_   4
#define DFF_    1024
#define NCLS_   80
#define NLAYERS_ 6
#define OFFC_   (NHEAD_*NLEV_*NPTS_*2)   // 192
#define ATTC_   (NHEAD_*NLEV_*NPTS_)     // 96

// ---------------- WMMA types ----------------
typedef __bf16 bf16_t;
typedef bf16_t v16bf __attribute__((ext_vector_type(16)));
typedef float  v8f   __attribute__((ext_vector_type(8)));
typedef unsigned int v4u __attribute__((ext_vector_type(4)));

union Frag {
    v16bf v;
    v4u   q[2];
};

__device__ __host__ __forceinline__ unsigned short f2bf(float f) {
    unsigned int u = __builtin_bit_cast(unsigned int, f);
    unsigned int r = (u + 0x7FFFu + ((u >> 16) & 1u)) >> 16;  // RNE
    return (unsigned short)r;
}

// issue one 16-byte async global->LDS transfer (ASYNCcnt-tracked)
__device__ __forceinline__ void async_cp16(unsigned lds_off, const unsigned short* g) {
    unsigned long long ga = (unsigned long long)(uintptr_t)g;
    asm volatile("global_load_async_to_lds_b128 %0, %1, off"
                 :: "v"(lds_off), "v"(ga) : "memory");
}

// =====================================================================
// bf16 WMMA GEMM: C[M,N] = act(A[M,Kp] @ Wt[Np,Kp]^T + bias[N])
//   A  : bf16, M x Kp, M % 64 == 0, Kp % 32 == 0 (zero-padded)
//   Wt : bf16, Np x Kp (weight transposed, zero-padded), Np % 128 == 0
// Block: 256 thr = 8 waves (2 x 4); block tile 64 x 128;
// wave tile 32 x 32 -> 4 x v_wmma_f32_16x16x32_bf16 per K-step.
// Double-buffered LDS staging with global_load_async_to_lds_b128:
// while computing tile t, tile t+1 is in flight (s_wait_asynccnt 3).
// Fragments read as 2 x ds_load_b128 each (layout-native, PITCH=40).
// =====================================================================
#define TMB 64
#define TNB 128
#define TKB 32
#define PITCH 40   // 32 + 8 pad ushorts -> 80B rows (multiple of 16B)

__global__ __launch_bounds__(256) void gemm_bf16_async(
    const unsigned short* __restrict__ A,
    const unsigned short* __restrict__ Wt,
    const float* __restrict__ bias,
    float* __restrict__ C,
    int M, int N, int Kp, int relu)
{
    __shared__ __align__(16) unsigned short sA[2 * TMB * PITCH];
    __shared__ __align__(16) unsigned short sB[2 * TNB * PITCH];

    const int tid  = threadIdx.x;
    const int lane = tid & 31;
    const int wave = tid >> 5;
    const int wm   = wave >> 2;          // 0..1 : 32-row group
    const int wn   = wave & 3;           // 0..3 : 32-col group
    const int m0   = blockIdx.x * TMB;
    const int n0   = blockIdx.y * TNB;
    const int hi   = lane >> 4;
    const int l15  = lane & 15;

    // ---- async staging assignments (all in-bounds: padded operands) ----
    const int ar = tid >> 2, as = tid & 3;                 // A: 64 rows x 4 segs
    const unsigned aLds = (unsigned)(uintptr_t)(sA + ar * PITCH + as * 8);
    const unsigned short* aGlb = A + (size_t)(m0 + ar) * Kp + as * 8;

    unsigned bLds[2];
    const unsigned short* bGlb[2];
    #pragma unroll
    for (int j = 0; j < 2; ++j) {                          // B: 128 rows x 4 segs
        int idx = tid * 2 + j;
        int br = idx >> 2, bs = idx & 3;
        bLds[j] = (unsigned)(uintptr_t)(sB + br * PITCH + bs * 8);
        bGlb[j] = Wt + (size_t)(n0 + br) * Kp + bs * 8;
    }

    const unsigned aStep = TMB * PITCH * 2;   // bytes per A ping-pong buffer
    const unsigned bStep = TNB * PITCH * 2;   // bytes per B ping-pong buffer

    v8f acc[2][2] = {};

    const int nK = Kp / TKB;

    // prologue: tile 0 -> buffer 0
    async_cp16(aLds,    aGlb);
    async_cp16(bLds[0], bGlb[0]);
    async_cp16(bLds[1], bGlb[1]);

    for (int t = 0; t < nK; ++t) {
        const int buf = t & 1;
        if (t + 1 < nK) {
            // issue tile t+1 into the other buffer, then wait only for tile t
            const unsigned nb = (unsigned)((t + 1) & 1);
            const int kn = (t + 1) * TKB;
            async_cp16(aLds    + nb * aStep, aGlb    + kn);
            async_cp16(bLds[0] + nb * bStep, bGlb[0] + kn);
            async_cp16(bLds[1] + nb * bStep, bGlb[1] + kn);
            asm volatile("s_wait_asynccnt 0x3" ::: "memory");
        } else {
            asm volatile("s_wait_asynccnt 0x0" ::: "memory");
        }
        __syncthreads();

        const unsigned short* sAb = sA + buf * (TMB * PITCH);
        const unsigned short* sBb = sB + buf * (TNB * PITCH);

        // A fragment: row = wm*32 + ai*16 + l15 ; K = {hi*8+e, 16+hi*8+e}
        Frag fa[2];
        #pragma unroll
        for (int ai = 0; ai < 2; ++ai) {
            const v4u* p = (const v4u*)(sAb + (wm*32 + ai*16 + l15) * PITCH + hi * 8);
            fa[ai].q[0] = p[0];     // +0B : K = hi*8 .. hi*8+7
            fa[ai].q[1] = p[2];     // +32B: K = 16+hi*8 .. 16+hi*8+7
        }
        // B fragment: col = wn*32 + bi*16 + l15 ; K = {hi*8+e, 16+hi*8+e}
        Frag fb[2];
        #pragma unroll
        for (int bi = 0; bi < 2; ++bi) {
            const v4u* p = (const v4u*)(sBb + (wn*32 + bi*16 + l15) * PITCH + hi * 8);
            fb[bi].q[0] = p[0];
            fb[bi].q[1] = p[2];
        }

        #pragma unroll
        for (int ai = 0; ai < 2; ++ai)
            #pragma unroll
            for (int bi = 0; bi < 2; ++bi)
                acc[ai][bi] = __builtin_amdgcn_wmma_f32_16x16x32_bf16(
                    false, fa[ai].v, false, fb[bi].v, (short)0,
                    acc[ai][bi], false, false);

        __syncthreads();   // reads of buf done before iter t+1 overwrites it
    }

    // D layout: VGPR r -> M = base + hi*8 + r ; lane l15 -> N
    #pragma unroll
    for (int bi = 0; bi < 2; ++bi) {
        const int gn = n0 + wn*32 + bi*16 + l15;
        if (gn < N) {
            const float bv = bias ? bias[gn] : 0.0f;
            #pragma unroll
            for (int ai = 0; ai < 2; ++ai) {
                const int gmb = m0 + wm*32 + ai*16 + hi*8;
                #pragma unroll
                for (int r = 0; r < 8; ++r) {
                    float v = acc[ai][bi][r] + bv;
                    if (relu) v = fmaxf(v, 0.0f);
                    C[(size_t)(gmb + r) * N + gn] = v;
                }
            }
        }
    }
}

// ---------------- operand conversion kernels ----------------
// A: (M x K) f32 -> (M x Kp) bf16, zero-padded K
__global__ void cvtA_k(const float* __restrict__ in, unsigned short* __restrict__ out,
                       int M, int K, int Kp) {
    int i = blockIdx.x * blockDim.x + threadIdx.x;
    if (i >= M * Kp) return;
    int m = i / Kp, k = i - m * Kp;
    out[i] = (k < K) ? f2bf(in[(size_t)m * K + k]) : (unsigned short)0;
}
// W: (K x N) f32 -> transposed (Np x Kp) bf16, zero-padded
__global__ void cvtWt_k(const float* __restrict__ in, unsigned short* __restrict__ out,
                        int K, int N, int Kp, int Np) {
    int i = blockIdx.x * blockDim.x + threadIdx.x;
    if (i >= Np * Kp) return;
    int n = i / Kp, k = i - n * Kp;
    out[i] = (k < K && n < N) ? f2bf(in[(size_t)k * N + n]) : (unsigned short)0;
}

// =====================================================================
// MHA: one wave per (query, head); lane = channel of DH=32.
// =====================================================================
__global__ __launch_bounds__(256) void mha_attn(
    const float* __restrict__ Q, const float* __restrict__ K,
    const float* __restrict__ V, float* __restrict__ O)
{
    const int q = blockIdx.x, b = blockIdx.y;
    const int h = threadIdx.x >> 5, lane = threadIdx.x & 31;

    const size_t qi = (((size_t)b * NQ_ + q) * NHEAD_ + h) * DH_ + lane;
    const float qv = Q[qi] * 0.17677669529663687f;  // 1/sqrt(32)

    float m = -1e30f, l = 0.0f, acc = 0.0f;
    for (int k = 0; k < NQ_; ++k) {
        const size_t ki = (((size_t)b * NQ_ + k) * NHEAD_ + h) * DH_ + lane;
        float s = qv * K[ki];
        #pragma unroll
        for (int off = 16; off > 0; off >>= 1)
            s += __shfl_xor(s, off, 32);
        float mn   = fmaxf(m, s);
        float corr = __expf(m - mn);
        float p    = __expf(s - mn);
        l   = l * corr + p;
        acc = acc * corr + p * V[ki];
        m = mn;
    }
    O[qi] = acc / l;
}

// =====================================================================
// LayerNorm(x + r) * g + b ; one 256-thread block per row. In-place safe.
// =====================================================================
__global__ __launch_bounds__(256) void layernorm_res(
    const float* __restrict__ x, const float* __restrict__ r,
    const float* __restrict__ g, const float* __restrict__ b,
    float* __restrict__ o)
{
    __shared__ float red[256];
    const int row = blockIdx.x, t = threadIdx.x;
    const size_t idx = (size_t)row * D_ + t;
    float v = x[idx] + r[idx];

    red[t] = v; __syncthreads();
    for (int s = 128; s > 0; s >>= 1) { if (t < s) red[t] += red[t + s]; __syncthreads(); }
    float mean = red[0] * (1.0f / D_);
    __syncthreads();
    float d = v - mean;
    red[t] = d * d; __syncthreads();
    for (int s = 128; s > 0; s >>= 1) { if (t < s) red[t] += red[t + s]; __syncthreads(); }
    float inv = rsqrtf(red[0] * (1.0f / D_) + 1e-5f);
    o[idx] = d * inv * g[t] + b[t];
}

// =====================================================================
// Deformable attention sampling. One wave per (query, head).
// =====================================================================
__global__ __launch_bounds__(256) void deform_sample(
    const float* __restrict__ value, const float* __restrict__ off,
    const float* __restrict__ aw, const float* __restrict__ refd,
    float* __restrict__ out)
{
    const int q = blockIdx.x, b = blockIdx.y;
    const int h = threadIdx.x >> 5, lane = threadIdx.x & 31;
    const size_t bq = (size_t)b * NQ_ + q;

    const float* al = aw + bq * ATTC_ + h * (NLEV_ * NPTS_);
    float w[NLEV_ * NPTS_];
    float mx = -1e30f;
    #pragma unroll
    for (int j = 0; j < NLEV_ * NPTS_; ++j) { w[j] = al[j]; mx = fmaxf(mx, w[j]); }
    float sum = 0.0f;
    #pragma unroll
    for (int j = 0; j < NLEV_ * NPTS_; ++j) { w[j] = __expf(w[j] - mx); sum += w[j]; }
    const float isum = 1.0f / sum;

    const float* rf = refd + bq * 4;
    const float cx = rf[0], cy = rf[1], rw = rf[2], rh = rf[3];
    const float* of = off + bq * OFFC_ + h * (NLEV_ * NPTS_ * 2);

    const int Hs[NLEV_] = {80, 40, 20};
    const int Ws[NLEV_] = {80, 40, 20};
    const int Ss[NLEV_] = {0, 6400, 8000};

    float acc = 0.0f;
    #pragma unroll
    for (int lv = 0; lv < NLEV_; ++lv) {
        const int Hl = Hs[lv], Wl = Ws[lv];
        const size_t vbase = ((size_t)b * LM_ + Ss[lv]) * D_ + h * DH_ + lane;
        #pragma unroll
        for (int pt = 0; pt < NPTS_; ++pt) {
            const float ox = of[(lv * NPTS_ + pt) * 2 + 0];
            const float oy = of[(lv * NPTS_ + pt) * 2 + 1];
            const float lx = cx + ox * (0.5f / NPTS_) * rw;
            const float ly = cy + oy * (0.5f / NPTS_) * rh;
            const float x = lx * Wl - 0.5f;
            const float y = ly * Hl - 0.5f;
            const float x0f = floorf(x), y0f = floorf(y);
            const float wx = x - x0f, wy = y - y0f;
            const int x0 = (int)x0f, y0 = (int)y0f;
            float s = 0.0f;
            #pragma unroll
            for (int c = 0; c < 4; ++c) {
                const int dx = c & 1, dy = c >> 1;
                const float cw = (dx ? wx : 1.0f - wx) * (dy ? wy : 1.0f - wy);
                const int xi = x0 + dx, yi = y0 + dy;
                const float valid = (xi >= 0 && xi < Wl && yi >= 0 && yi < Hl) ? 1.0f : 0.0f;
                const int xc = min(max(xi, 0), Wl - 1);
                const int yc = min(max(yi, 0), Hl - 1);
                const float gsv = value[vbase + (size_t)(yc * Wl + xc) * D_];
                s += gsv * (cw * valid);
            }
            acc += s * (w[lv * NPTS_ + pt] * isum);
        }
    }
    out[bq * D_ + h * DH_ + lane] = acc;
}

// ---------------- elementwise kernels ----------------
__global__ void add_k(const float* a, const float* b, float* o, int n) {
    int i = blockIdx.x * blockDim.x + threadIdx.x;
    if (i < n) o[i] = a[i] + b[i];
}
__global__ void sigmoid_k(const float* x, float* o, int n) {
    int i = blockIdx.x * blockDim.x + threadIdx.x;
    if (i < n) o[i] = 1.0f / (1.0f + __expf(-x[i]));
}
__global__ void bbox_update_k(float* ref, const float* delta, int n) {
    int i = blockIdx.x * blockDim.x + threadIdx.x;
    if (i < n) {
        float x   = fminf(fmaxf(ref[i], 0.0f), 1.0f);
        float num = fmaxf(x, 1e-5f);
        float den = fmaxf(1.0f - x, 1e-5f);
        float inv = __logf(num / den);
        ref[i] = 1.0f / (1.0f + __expf(-(delta[i] + inv)));
    }
}

// ---------------- host-side helper ----------------
static inline void gemm_bf(const unsigned short* A, const unsigned short* Wt,
                           const float* bias, float* C,
                           int M, int N, int Kp, int Np, int relu, hipStream_t s) {
    dim3 grid(M / TMB, Np / TNB);
    gemm_bf16_async<<<grid, 256, 0, s>>>(A, Wt, bias, C, M, N, Kp, relu);
}

extern "C" void kernel_launch(void* const* d_in, const int* in_sizes, int n_in,
                              void* d_out, int out_size, void* d_ws, size_t ws_size,
                              hipStream_t stream) {
    const float* tgt     = (const float*)d_in[0];
    const float* refu    = (const float*)d_in[1];
    const float* memory  = (const float*)d_in[2];
    const float* sa_wq   = (const float*)d_in[3];
    const float* sa_wk   = (const float*)d_in[4];
    const float* sa_wv   = (const float*)d_in[5];
    const float* sa_wo   = (const float*)d_in[6];
    const float* val_w   = (const float*)d_in[7];
    const float* out_w   = (const float*)d_in[8];
    const float* bb_w1   = (const float*)d_in[9];
    const float* bb_w2   = (const float*)d_in[10];
    const float* sa_bq   = (const float*)d_in[11];
    const float* sa_bk   = (const float*)d_in[12];
    const float* sa_bv   = (const float*)d_in[13];
    const float* sa_bo   = (const float*)d_in[14];
    const float* ln1_b   = (const float*)d_in[15];
    const float* ln2_b   = (const float*)d_in[16];
    const float* ln3_b   = (const float*)d_in[17];
    const float* val_b   = (const float*)d_in[18];
    const float* out_b   = (const float*)d_in[19];
    const float* ffn_b2  = (const float*)d_in[20];
    const float* bb_b1   = (const float*)d_in[21];
    const float* bb_b2   = (const float*)d_in[22];
    const float* ln1_g   = (const float*)d_in[23];
    const float* ln2_g   = (const float*)d_in[24];
    const float* ln3_g   = (const float*)d_in[25];
    const float* off_w   = (const float*)d_in[26];
    const float* off_b   = (const float*)d_in[27];
    const float* attw_w  = (const float*)d_in[28];
    const float* attw_b  = (const float*)d_in[29];
    const float* ffn_w1  = (const float*)d_in[30];
    const float* ffn_b1  = (const float*)d_in[31];
    const float* ffn_w2  = (const float*)d_in[32];
    const float* bb_w3   = (const float*)d_in[33];
    const float* bb_b3   = (const float*)d_in[34];
    const float* sc_w    = (const float*)d_in[35];
    const float* sc_b    = (const float*)d_in[36];
    const float* qp_w1   = (const float*)d_in[37];
    const float* qp_b1   = (const float*)d_in[38];
    const float* qp_w2   = (const float*)d_in[39];
    const float* qp_b2   = (const float*)d_in[40];
    (void)in_sizes; (void)n_in; (void)out_size; (void)ws_size;

    const size_t MQ = (size_t)BS_ * NQ_;   // 4800 (75 * 64)
    const size_t MV = (size_t)BS_ * LM_;   // 134400 (2100 * 64)
    const int iMQ = (int)MQ, iMV = (int)MV;
    const int ETHR = 256;
    const int nQ4 = (int)(MQ * 4);
    const int nQD = (int)(MQ * D_);

    // -------- workspace (byte bump allocator, 256B aligned) --------
    char* wsp = (char*)d_ws;
    auto takeB = [&](size_t bytes) {
        char* r = wsp; wsp += (bytes + 255) & ~(size_t)255; return r;
    };
    float* bOut = (float*)takeB(MQ * D_ * 4);
    float* bPos = (float*)takeB(MQ * D_ * 4);
    float* bQin = (float*)takeB(MQ * D_ * 4);
    float* bQ   = (float*)takeB(MQ * D_ * 4);
    float* bK   = (float*)takeB(MQ * D_ * 4);
    float* bV   = (float*)takeB(MQ * D_ * 4);
    float* bT   = (float*)takeB(MQ * D_ * 4);
    float* bH   = (float*)takeB(MQ * DFF_ * 4);
    float* bOff = (float*)takeB(MQ * OFFC_ * 4);
    float* bAw  = (float*)takeB(MQ * ATTC_ * 4);
    float* bRef = (float*)takeB(MQ * 4 * 4);
    float* bDel = (float*)takeB(MQ * 4 * 4);
    float* bVal = (float*)takeB(MV * D_ * 4);
    unsigned short* bMem   = (unsigned short*)takeB(MV * D_ * 2);   // memory in bf16
    unsigned short* bAbf   = (unsigned short*)takeB(MQ * DFF_ * 2); // activation scratch
    unsigned short* wArena = (unsigned short*)takeB(20u * 1024 * 1024);
    size_t wOff = 0;

    // convert+transpose one weight matrix into the bf16 arena
    auto cvtW = [&](const float* w, int K, int N, int& KpO, int& NpO)
            -> const unsigned short* {
        int Kp = (K + 31) & ~31;
        int Np = (N + 127) & ~127;
        unsigned short* dst = wArena + wOff;
        wOff += (size_t)Kp * Np;
        int tot = Kp * Np;
        cvtWt_k<<<(tot + ETHR - 1) / ETHR, ETHR, 0, stream>>>(w, dst, K, N, Kp, Np);
        KpO = Kp; NpO = Np;
        return dst;
    };
    // convert one activation matrix into bAbf (stream-ordered reuse)
    auto cvtA = [&](const float* a, int M, int K) -> const unsigned short* {
        int Kp = (K + 31) & ~31;
        int tot = M * Kp;
        cvtA_k<<<(tot + ETHR - 1) / ETHR, ETHR, 0, stream>>>(a, bAbf, M, K, Kp);
        return bAbf;
    };

    // -------- one-time conversions --------
    {
        int tot = iMV * D_;
        cvtA_k<<<(tot + ETHR - 1) / ETHR, ETHR, 0, stream>>>(memory, bMem, iMV, D_, D_);
    }
    int kqp1, nqp1, kqp2, nqp2, ksc, nsc;
    const unsigned short* qp1t = cvtW(qp_w1, 4, 2 * D_, kqp1, nqp1);
    const unsigned short* qp2t = cvtW(qp_w2, 2 * D_, D_, kqp2, nqp2);
    const unsigned short* sct  = cvtW(sc_w + (size_t)(NLAYERS_ - 1) * D_ * NCLS_,
                                      D_, NCLS_, ksc, nsc);

    hipMemcpyAsync(bOut, tgt, MQ * D_ * sizeof(float),
                   hipMemcpyDeviceToDevice, stream);
    sigmoid_k<<<(nQ4 + ETHR - 1) / ETHR, ETHR, 0, stream>>>(refu, bRef, nQ4);

    int kp, np;
    for (int i = 0; i < NLAYERS_; ++i) {
        // ----- query-pos embedding: pos = mlp2(ref_d) -----
        gemm_bf(cvtA(bRef, iMQ, 4), qp1t, qp_b1, bH, iMQ, 2 * D_, kqp1, nqp1, 1, stream);
        gemm_bf(cvtA(bH, iMQ, 2 * D_), qp2t, qp_b2, bPos, iMQ, D_, kqp2, nqp2, 0, stream);

        // ----- self-attention -----
        add_k<<<(nQD + ETHR - 1) / ETHR, ETHR, 0, stream>>>(bOut, bPos, bQin, nQD);
        const unsigned short* wqT = cvtW(sa_wq + (size_t)i * D_ * D_, D_, D_, kp, np);
        const unsigned short* wkT = cvtW(sa_wk + (size_t)i * D_ * D_, D_, D_, kp, np);
        const unsigned short* wvT = cvtW(sa_wv + (size_t)i * D_ * D_, D_, D_, kp, np);
        const unsigned short* aQ = cvtA(bQin, iMQ, D_);
        gemm_bf(aQ, wqT, sa_bq + (size_t)i * D_, bQ, iMQ, D_, kp, np, 0, stream);
        gemm_bf(aQ, wkT, sa_bk + (size_t)i * D_, bK, iMQ, D_, kp, np, 0, stream);
        gemm_bf(cvtA(bOut, iMQ, D_), wvT, sa_bv + (size_t)i * D_, bV, iMQ, D_, kp, np, 0, stream);
        mha_attn<<<dim3(NQ_, BS_), 256, 0, stream>>>(bQ, bK, bV, bT);
        const unsigned short* woT = cvtW(sa_wo + (size_t)i * D_ * D_, D_, D_, kp, np);
        gemm_bf(cvtA(bT, iMQ, D_), woT, sa_bo + (size_t)i * D_, bQ, iMQ, D_, kp, np, 0, stream);
        layernorm_res<<<iMQ, 256, 0, stream>>>(bOut, bQ,
            ln1_g + (size_t)i * D_, ln1_b + (size_t)i * D_, bOut);

        // ----- deformable cross-attention -----
        add_k<<<(nQD + ETHR - 1) / ETHR, ETHR, 0, stream>>>(bOut, bPos, bQin, nQD);
        int kpo, npo, kpa, npa;
        const unsigned short* ofT = cvtW(off_w  + (size_t)i * D_ * OFFC_, D_, OFFC_, kpo, npo);
        const unsigned short* awT = cvtW(attw_w + (size_t)i * D_ * ATTC_, D_, ATTC_, kpa, npa);
        const unsigned short* aQ2 = cvtA(bQin, iMQ, D_);
        gemm_bf(aQ2, ofT, off_b  + (size_t)i * OFFC_, bOff, iMQ, OFFC_, kpo, npo, 0, stream);
        gemm_bf(aQ2, awT, attw_b + (size_t)i * ATTC_, bAw,  iMQ, ATTC_, kpa, npa, 0, stream);
        const unsigned short* vwT = cvtW(val_w + (size_t)i * D_ * D_, D_, D_, kp, np);
        gemm_bf(bMem, vwT, val_b + (size_t)i * D_, bVal, iMV, D_, kp, np, 0, stream);
        deform_sample<<<dim3(NQ_, BS_), 256, 0, stream>>>(bVal, bOff, bAw, bRef, bT);
        const unsigned short* owT = cvtW(out_w + (size_t)i * D_ * D_, D_, D_, kp, np);
        gemm_bf(cvtA(bT, iMQ, D_), owT, out_b + (size_t)i * D_, bQ, iMQ, D_, kp, np, 0, stream);
        layernorm_res<<<iMQ, 256, 0, stream>>>(bOut, bQ,
            ln2_g + (size_t)i * D_, ln2_b + (size_t)i * D_, bOut);

        // ----- FFN -----
        int kpf1, npf1, kpf2, npf2;
        const unsigned short* f1T = cvtW(ffn_w1 + (size_t)i * D_ * DFF_, D_, DFF_, kpf1, npf1);
        gemm_bf(cvtA(bOut, iMQ, D_), f1T, ffn_b1 + (size_t)i * DFF_, bH, iMQ, DFF_, kpf1, npf1, 1, stream);
        const unsigned short* f2T = cvtW(ffn_w2 + (size_t)i * DFF_ * D_, DFF_, D_, kpf2, npf2);
        gemm_bf(cvtA(bH, iMQ, DFF_), f2T, ffn_b2 + (size_t)i * D_, bT, iMQ, D_, kpf2, npf2, 0, stream);
        layernorm_res<<<iMQ, 256, 0, stream>>>(bOut, bT,
            ln3_g + (size_t)i * D_, ln3_b + (size_t)i * D_, bOut);

        // ----- bbox refinement -----
        const unsigned short* b1T = cvtW(bb_w1 + (size_t)i * D_ * D_, D_, D_, kp, np);
        gemm_bf(cvtA(bOut, iMQ, D_), b1T, bb_b1 + (size_t)i * D_, bH, iMQ, D_, kp, np, 1, stream);
        const unsigned short* b2T = cvtW(bb_w2 + (size_t)i * D_ * D_, D_, D_, kp, np);
        gemm_bf(cvtA(bH, iMQ, D_), b2T, bb_b2 + (size_t)i * D_, bT, iMQ, D_, kp, np, 1, stream);
        int kp3, np3;
        const unsigned short* b3T = cvtW(bb_w3 + (size_t)i * D_ * 4, D_, 4, kp3, np3);
        gemm_bf(cvtA(bT, iMQ, D_), b3T, bb_b3 + (size_t)i * 4, bDel, iMQ, 4, kp3, np3, 0, stream);
        bbox_update_k<<<(nQ4 + ETHR - 1) / ETHR, ETHR, 0, stream>>>(bRef, bDel, nQ4);
    }

    // outputs: [ref_d (B*NQ*4), logits (B*NQ*NCLS)]
    float* out_f = (float*)d_out;
    hipMemcpyAsync(out_f, bRef, MQ * 4 * sizeof(float),
                   hipMemcpyDeviceToDevice, stream);
    gemm_bf(cvtA(bOut, iMQ, D_), sct, sc_b + (size_t)(NLAYERS_ - 1) * NCLS_,
            out_f + MQ * 4, iMQ, NCLS_, ksc, nsc, 0, stream);
}
